// WeDeepseekV3MLP_11613591568694
// MI455X (gfx1250) — 8-bit matrix operand order repaired and checked by reading, compiled, not run on hardware
//
#include <hip/hip_runtime.h>
#include <stdint.h>


typedef int   v16i __attribute__((ext_vector_type(16)));
typedef float v8f  __attribute__((ext_vector_type(8)));
typedef int   b128_t __attribute__((vector_size(16)));

#define FP8_GROUP 128
#define FP8_MAXV  448.0f

#if defined(__has_builtin)
#if __has_builtin(__builtin_amdgcn_global_load_async_to_lds_b128)
#define HAVE_ASYNC_BUILTIN 1
#endif
#if __has_builtin(__builtin_amdgcn_s_wait_asynccnt)
#define HAVE_WAIT_BUILTIN 1
#endif
#endif

#if defined(HAVE_ASYNC_BUILTIN)
#define ASYNC_COPY_B128(g, l)                                              \
  __builtin_amdgcn_global_load_async_to_lds_b128(                          \
      (__attribute__((address_space(1))) b128_t*)(uintptr_t)(g),           \
      (__attribute__((address_space(3))) b128_t*)(uintptr_t)(l), 0, 0)
#else
#define ASYNC_COPY_B128(g, l)                                              \
  asm volatile("global_load_async_to_lds_b128 %0, %1, off"                 \
               ::"v"((unsigned)(uintptr_t)(l)), "v"((const void*)(g))      \
               : "memory")
#endif

#if defined(HAVE_WAIT_BUILTIN)
#define WAIT_ASYNC(n) __builtin_amdgcn_s_wait_asynccnt(n)
#else
#define WAIT_ASYNC(n) asm volatile("s_wait_asynccnt %0" ::"i"(n) : "memory")
#endif

__device__ __forceinline__ unsigned int f32_to_e4m3(float x) {
  unsigned int u = __float_as_uint(x);
  unsigned int sign = (u >> 24) & 0x80u;
  float ax = fminf(fabsf(x), FP8_MAXV);
  unsigned int code;
  if (ax < 0.015625f) {
    code = (unsigned int)rintf(ax * 512.0f);
  } else {
    unsigned int au = __float_as_uint(ax);
    au += 0x0007FFFFu + ((au >> 20) & 1u);
    unsigned int e = (au >> 23) - 127u + 7u;
    unsigned int m = (au >> 20) & 7u;
    code = (e << 3) | m;
    if (code > 0x7Eu) code = 0x7Eu;
  }
  return sign | code;
}

__device__ __forceinline__ unsigned int pack4_e4m3(float4 v, float inv) {
  return  f32_to_e4m3(v.x * inv)
       | (f32_to_e4m3(v.y * inv) << 8)
       | (f32_to_e4m3(v.z * inv) << 16)
       | (f32_to_e4m3(v.w * inv) << 24);
}

__global__ __launch_bounds__(256) void quant_token_128(
    const float* __restrict__ x, uint8_t* __restrict__ q,
    float* __restrict__ s, int ngroups) {
  const int g    = blockIdx.x * 8 + (threadIdx.x >> 5);
  const int lane = threadIdx.x & 31;
  if (g >= ngroups) return;
  float4 v = ((const float4*)x)[(size_t)g * 32 + lane];
  float m = fmaxf(fmaxf(fabsf(v.x), fabsf(v.y)), fmaxf(fabsf(v.z), fabsf(v.w)));
#pragma unroll
  for (int o = 16; o > 0; o >>= 1) m = fmaxf(m, __shfl_xor(m, o, 32));
  m = fmaxf(m, 1e-4f);
  const float scale = m * (1.0f / FP8_MAXV);
  const float inv   = FP8_MAXV / m;
  ((unsigned int*)q)[(size_t)g * 32 + lane] = pack4_e4m3(v, inv);
  if (lane == 0) s[g] = scale;
}

__global__ __launch_bounds__(256) void quant_block_128x128(
    const float* __restrict__ w, uint8_t* __restrict__ wq,
    float* __restrict__ wscale, int N, int K) {
  const int kb = blockIdx.x, nb = blockIdx.y;
  const int KG = K / FP8_GROUP;
  const int tid = threadIdx.x;
  __shared__ float red[256];
  __shared__ float s_inv;

  float4 vals[16];
  float m = 0.0f;
#pragma unroll
  for (int it = 0; it < 16; ++it) {
    int f = tid + it * 256;
    int row = f >> 5, c4 = f & 31;
    const float4* src = (const float4*)(w + (size_t)(nb * 128 + row) * K + kb * 128);
    float4 v = src[c4];
    vals[it] = v;
    m = fmaxf(m, fmaxf(fmaxf(fabsf(v.x), fabsf(v.y)), fmaxf(fabsf(v.z), fabsf(v.w))));
  }
  red[tid] = m;
  __syncthreads();
  for (int st = 128; st > 0; st >>= 1) {
    if (tid < st) red[tid] = fmaxf(red[tid], red[tid + st]);
    __syncthreads();
  }
  if (tid == 0) {
    float amax = fmaxf(red[0], 1e-4f);
    s_inv = FP8_MAXV / amax;
    wscale[(size_t)nb * KG + kb] = amax * (1.0f / FP8_MAXV);
  }
  __syncthreads();
  const float inv = s_inv;
#pragma unroll
  for (int it = 0; it < 16; ++it) {
    int f = tid + it * 256;
    int row = f >> 5, c4 = f & 31;
    *(unsigned int*)(wq + (size_t)(nb * 128 + row) * K + kb * 128 + c4 * 4) =
        pack4_e4m3(vals[it], inv);
  }
}

template <bool FUSED>
__device__ __forceinline__ void issue_group_loads(
    const uint8_t* __restrict__ Aq, const uint8_t* __restrict__ B0q,
    const uint8_t* __restrict__ B1q, uint8_t* lA, uint8_t* lB0, uint8_t* lB1,
    int tid, int bm, int bn, int K, size_t kOff) {
#pragma unroll
  for (int it = 0; it < 4; ++it) {
    int idx = tid + it * 256;
    int row = idx >> 3, ch = idx & 7;
    ASYNC_COPY_B128(Aq + (size_t)(bm + row) * K + kOff + ch * 16, lA + idx * 16);
  }
#pragma unroll
  for (int it = 0; it < 2; ++it) {
    int idx = tid + it * 256;
    int row = idx >> 3, ch = idx & 7;
    ASYNC_COPY_B128(B0q + (size_t)(bn + row) * K + kOff + ch * 16, lB0 + idx * 16);
    if (FUSED)
      ASYNC_COPY_B128(B1q + (size_t)(bn + row) * K + kOff + ch * 16, lB1 + idx * 16);
  }
}

template <bool FUSED>
__global__ __launch_bounds__(256) void fp8_gemm_128x64(
    const uint8_t* __restrict__ Aq,  const float* __restrict__ As,
    const uint8_t* __restrict__ B0q, const float* __restrict__ B0s,
    const uint8_t* __restrict__ B1q, const float* __restrict__ B1s,
    float* __restrict__ Out, int M, int N, int K) {
  const int KG = K / FP8_GROUP;
  __shared__ uint8_t lA[2][128 * 128];
  __shared__ uint8_t lB0[2][64 * 128];
  __shared__ uint8_t lB1[2][FUSED ? 64 * 128 : 16];
  __shared__ float   lsx[2][128];

  const int tid  = threadIdx.x;
  const int lane = tid & 31, wid = tid >> 5;
  const int bm = blockIdx.y * 128, bn = blockIdx.x * 64;
  const int wm = (wid & 3) * 32,   wn = (wid >> 2) * 32;
  const int nb   = bn >> 7;
  const int half = lane >> 4, l16 = lane & 15;

  union F8  { v8f  v; float f[8]; };
  union F16 { v16i v; int   i[16]; };
  F8 accG[2][2], accU[2][2];
#pragma unroll
  for (int a = 0; a < 2; ++a)
#pragma unroll
    for (int b = 0; b < 2; ++b)
#pragma unroll
      for (int i = 0; i < 8; ++i) { accG[a][b].f[i] = 0.f; accU[a][b].f[i] = 0.f; }

  issue_group_loads<FUSED>(Aq, B0q, B1q, lA[0], lB0[0], lB1[0], tid, bm, bn, K, 0);
  if (tid < 128) lsx[0][tid] = As[(size_t)(bm + tid) * KG];

  for (int kg = 0; kg < KG; ++kg) {
    const int p = kg & 1;

    if (kg + 1 < KG) {
      issue_group_loads<FUSED>(Aq, B0q, B1q, lA[1 - p], lB0[1 - p], lB1[1 - p],
                               tid, bm, bn, K, (size_t)(kg + 1) * FP8_GROUP);
      if (tid < 128) lsx[1 - p][tid] = As[(size_t)(bm + tid) * KG + (kg + 1)];
      WAIT_ASYNC(FUSED ? 8 : 6);
    } else {
      WAIT_ASYNC(0);
    }
    __syncthreads();

    const uint8_t* cA  = lA[p];
    const uint8_t* cB0 = lB0[p];
    const uint8_t* cB1 = lB1[p];
    const float*   csx = lsx[p];

    const float sw0 = B0s[(size_t)nb * KG + kg];
    const float sw1 = FUSED ? B1s[(size_t)nb * KG + kg] : 0.f;

    F16 af[2];
    float rs[2][8];
#pragma unroll
    for (int mt = 0; mt < 2; ++mt) {
      const uint8_t* base = cA + (wm + mt * 16 + l16) * 128 + half * 16;
#pragma unroll
      for (int g = 0; g < 4; ++g) {
        int4 t = *(const int4*)(base + g * 32);
        af[mt].i[4 * g + 0] = t.x; af[mt].i[4 * g + 1] = t.y;
        af[mt].i[4 * g + 2] = t.z; af[mt].i[4 * g + 3] = t.w;
      }
      const int rbase = wm + mt * 16 + half * 8;
#pragma unroll
      for (int i = 0; i < 8; ++i) rs[mt][i] = csx[rbase + i];
    }

#pragma unroll
    for (int nt = 0; nt < 2; ++nt) {
      const int col = wn + nt * 16 + l16;
      F16 bf;
      const uint8_t* bb = cB0 + col * 128 + half * 16;
#pragma unroll
      for (int q = 0; q < 4; ++q) {
        int4 t = *(const int4*)(bb + q * 32);
        bf.i[4 * q + 0] = t.x; bf.i[4 * q + 1] = t.y;
        bf.i[4 * q + 2] = t.z; bf.i[4 * q + 3] = t.w;
      }
#pragma unroll
      for (int mt = 0; mt < 2; ++mt) {
        v8f z = {};
        F8 d;
        d.v = __builtin_amdgcn_wmma_f32_16x16x128_fp8_fp8(
            af[mt].v, bf.v, (short)0, z, false, false);
#pragma unroll
        for (int i = 0; i < 8; ++i)
          accG[mt][nt].f[i] = fmaf(d.f[i], rs[mt][i] * sw0, accG[mt][nt].f[i]);
      }
      if (FUSED) {
        const uint8_t* bb1 = cB1 + col * 128 + half * 16;
#pragma unroll
        for (int q = 0; q < 4; ++q) {
          int4 t = *(const int4*)(bb1 + q * 32);
          bf.i[4 * q + 0] = t.x; bf.i[4 * q + 1] = t.y;
          bf.i[4 * q + 2] = t.z; bf.i[4 * q + 3] = t.w;
        }
#pragma unroll
        for (int mt = 0; mt < 2; ++mt) {
          v8f z = {};
          F8 d;
          d.v = __builtin_amdgcn_wmma_f32_16x16x128_fp8_fp8(
              af[mt].v, bf.v, (short)0, z, false, false);
#pragma unroll
          for (int i = 0; i < 8; ++i)
            accU[mt][nt].f[i] = fmaf(d.f[i], rs[mt][i] * sw1, accU[mt][nt].f[i]);
        }
      }
    }
    __syncthreads();
  }

#pragma unroll
  for (int mt = 0; mt < 2; ++mt)
#pragma unroll
    for (int nt = 0; nt < 2; ++nt)
#pragma unroll
      for (int i = 0; i < 8; ++i) {
        const int row = bm + wm + mt * 16 + half * 8 + i;
        const int col = bn + wn + nt * 16 + l16;
        if (FUSED) {
          float g = accG[mt][nt].f[i], u = accU[mt][nt].f[i];
          Out[(size_t)row * N + col] = (g / (1.0f + __expf(-g))) * u;
        } else {
          Out[(size_t)row * N + col] = accG[mt][nt].f[i];
        }
      }
}

extern "C" void kernel_launch(void* const* d_in, const int* in_sizes, int n_in,
                              void* d_out, int out_size, void* d_ws, size_t ws_size,
                              hipStream_t stream) {
  const float* x      = (const float*)d_in[0];
  const float* gate_w = (const float*)d_in[1];
  const float* up_w   = (const float*)d_in[2];
  const float* down_w = (const float*)d_in[3];
  float* out = (float*)d_out;

  const int M = 2048, H = 4096, I = 11008;
  const int HG = H / FP8_GROUP;
  const int IG = I / FP8_GROUP;

  uint8_t* ws = (uint8_t*)d_ws;
  size_t off = 0;
  auto alloc = [&](size_t bytes) -> void* {
    void* p = ws + off;
    off += (bytes + 255) & ~(size_t)255;
    return p;
  };
  uint8_t* xq = (uint8_t*)alloc((size_t)M * H);
  float*   xs = (float*)  alloc((size_t)M * HG * 4);
  uint8_t* gq = (uint8_t*)alloc((size_t)I * H);
  float*   gs = (float*)  alloc((size_t)IG * HG * 4);
  uint8_t* uq = (uint8_t*)alloc((size_t)I * H);
  float*   us = (float*)  alloc((size_t)IG * HG * 4);
  uint8_t* dq = (uint8_t*)alloc((size_t)H * I);
  float*   ds = (float*)  alloc((size_t)HG * IG * 4);
  float*   h  = (float*)  alloc((size_t)M * I * 4);
  uint8_t* hq = (uint8_t*)alloc((size_t)M * I);
  float*   hs = (float*)  alloc((size_t)M * IG * 4);

  quant_token_128<<<(M * HG) / 8, 256, 0, stream>>>(x, xq, xs, M * HG);
  quant_block_128x128<<<dim3(HG, IG), 256, 0, stream>>>(gate_w, gq, gs, I, H);
  quant_block_128x128<<<dim3(HG, IG), 256, 0, stream>>>(up_w,   uq, us, I, H);
  quant_block_128x128<<<dim3(IG, HG), 256, 0, stream>>>(down_w, dq, ds, H, I);
  fp8_gemm_128x64<true><<<dim3(I / 64, M / 128), 256, 0, stream>>>(
      xq, xs, gq, gs, uq, us, h, M, I, H);
  quant_token_128<<<(M * IG) / 8, 256, 0, stream>>>(h, hq, hs, M * IG);
  fp8_gemm_128x64<false><<<dim3(H / 64, M / 128), 256, 0, stream>>>(
      hq, hs, dq, ds, nullptr, nullptr, out, M, H, I);
}
